// RegressionGRU_7447473292113
// MI455X (gfx1250) — compile-verified, matrix operand order checked by reading, not run on hardware
//
#include <hip/hip_runtime.h>
#include <hip/hip_bf16.h>

// ---------------------------------------------------------------------------
// Problem constants (from the reference)
// ---------------------------------------------------------------------------
constexpr int kB  = 512;   // batch
constexpr int kT  = 512;   // time steps
constexpr int kI  = 64;    // input features (layer 0)
constexpr int kH  = 256;   // hidden
constexpr int kG3 = 3 * kH;         // 768 gate outputs (r,z,n)
constexpr int kK  = kH + kI;        // 320 fused K dim: [h | x_t]
constexpr int kAPAD  = 328;         // padded K stride for LDS A buffer (mult of 8)
constexpr int kBTILE = 16;          // batch rows per workgroup
constexpr int kTHREADS = 512;       // 16 wave32 per WG

typedef __attribute__((ext_vector_type(16))) __bf16 v16bf;
typedef __attribute__((ext_vector_type(8)))  float  v8f;
typedef __attribute__((ext_vector_type(4)))  unsigned int u32x4;  // trivial 128-bit

__device__ __forceinline__ float sigmoidf_(float v) {
    return 1.0f / (1.0f + __expf(-v));
}

// Force a 128-bit GLOBAL (address_space 1) load even when the pointer's
// provenance is hidden from addrspace inference by the anti-hoist asm.
// (Global aperture is identity-mapped on AMDGPU, so the cast is exact;
// this avoids FLAT loads, which tick DScnt and occupy the LDS issue path.)
// u32x4 is a builtin ext-vector, so loading through an AS(1) reference needs
// no constructor (unlike HIP's uint4 class) and compiles on host + device.
__device__ __forceinline__ u32x4 ldg_b128(const __bf16* p) {
    const __attribute__((address_space(1))) u32x4* gp =
        (const __attribute__((address_space(1))) u32x4*)(unsigned long long)p;
    return *gp;
}

// ---------------------------------------------------------------------------
// Prep: fuse [w_hh0 | w_ih0] -> bf16 wc[768][320], K-contiguous per output row
// (this is exactly the column-major-in-N layout the WMMA B fragment wants)
// ---------------------------------------------------------------------------
__global__ void gru_prep_weights(const float* __restrict__ w_ih,
                                 const float* __restrict__ w_hh,
                                 __bf16* __restrict__ wc) {
    int idx = blockIdx.x * blockDim.x + threadIdx.x;
    if (idx >= kG3 * kK) return;
    int n = idx / kK;
    int k = idx - n * kK;
    float v = (k < kH) ? w_hh[n * kH + k] : w_ih[n * kI + (k - kH)];
    wc[idx] = (__bf16)v;
}

// ---------------------------------------------------------------------------
// Layer-0 GRU over all T steps + final linear head.
// grid = 32 WGs (one 16-row batch tile each), block = 512 threads (16 waves).
// Wave w owns h-columns [16w, 16w+16): computes r,z,n for those columns and
// updates its h fragment (v8f, the WMMA C/D layout) fully in registers.
// Weights stream from L2 every step (anti-hoist asm prevents the compiler
// from register-pinning 240 VGPRs of weights and spilling to scratch).
// ---------------------------------------------------------------------------
__global__ __launch_bounds__(kTHREADS) void gru0_kernel(
    const float* __restrict__ x,       // [B, T, I] fp32
    const __bf16* __restrict__ wc,     // [768][320] bf16 fused weights
    const float* __restrict__ b_ih,    // [768]
    const float* __restrict__ b_hh,    // [768]
    const float* __restrict__ w_lin,   // [256]
    const float* __restrict__ b_lin,   // [1]
    float* __restrict__ out)           // [2B]
{
    __shared__ __align__(16) __bf16 abuf[kBTILE * kAPAD];  // A: [16 rows][h | x_t]
    __shared__ float hfin[kBTILE * kH];                    // final h (fp32) for head

    const int tid  = threadIdx.x;
    const int wave = tid >> 5;       // 0..15
    const int lane = tid & 31;
    const int l16  = lane & 15;
    const int kh   = lane >> 4;      // K-half / M-half select per ISA layouts
    const int rowbase = blockIdx.x * kBTILE;

    // This wave's h-column (= N column of its gate tiles), per D layout lane%16
    const int col = wave * 16 + l16;               // 0..255
    const float br  = b_ih[col]        + b_hh[col];
    const float bz  = b_ih[kH  + col]  + b_hh[kH  + col];
    const float bnx = b_ih[2*kH + col];            // x-side bias of n gate
    const float bnh = b_hh[2*kH + col];            // h-side bias of n gate

    // B-fragment base pointers for gates r / z / n (lane-divergent -> VGPRs)
    const __bf16* wr0 = wc + (size_t)(col)          * kK + kh * 8;
    const __bf16* wz0 = wc + (size_t)(kH   + col)   * kK + kh * 8;
    const __bf16* wn0 = wc + (size_t)(2*kH + col)   * kK + kh * 8;

    v8f h = {};   // h fragment: element e -> row M = e + 8*kh, col = col

    for (int t = 0; t < kT; ++t) {
        // Anti-hoist: redefine the weight pointers each iteration so the
        // B-fragment loads cannot be moved out of the t-loop (which would
        // pin 240 VGPRs of loop-invariant weights and spill to scratch).
        const __bf16* wr = wr0;
        const __bf16* wz = wz0;
        const __bf16* wn = wn0;
        asm volatile("" : "+v"(wr), "+v"(wz), "+v"(wn));

        // --- scatter current h (bf16) into A buffer cols [0,256) ------------
        #pragma unroll
        for (int e = 0; e < 8; ++e) {
            int m = e + kh * 8;
            abuf[m * kAPAD + col] = (__bf16)h[e];
        }
        // --- stage x_t into A buffer cols [256,320): 1024 elems, 512 thr ----
        {
            int idx = tid;
            #pragma unroll
            for (int rep = 0; rep < 2; ++rep) {
                int r = idx >> 6;            // 0..15
                int c = idx & 63;            // 0..63
                const float* xp = &x[((size_t)(rowbase + r) * kT + t) * kI + c];
                abuf[r * kAPAD + kH + c] = (__bf16)(*xp);
                if (t + 1 < kT) __builtin_prefetch(xp + kI, 0, 1); // next step
                idx += kTHREADS;
            }
        }
        __syncthreads();

        // --- gates = [h|x] @ W^T : K loop, bf16 WMMA, f32 accumulate --------
        v8f cr = {}, cz = {}, cnh = {}, cnx = {};
        #pragma unroll
        for (int kk = 0; kk < kK; kk += 32) {
            // A fragment (16x32 bf16): lane holds row l16, two 8-elem chunks
            v16bf a;
            const __bf16* ap = &abuf[l16 * kAPAD + kk + kh * 8];
            ((u32x4*)&a)[0] = *(const u32x4*)(ap);
            ((u32x4*)&a)[1] = *(const u32x4*)(ap + 16);

            // B fragments (32x16 bf16): lane holds column col, same K chunks.
            // Explicit global (AS1) b128 loads -> global_load_b128, LOADcnt only.
            v16bf fr, fz, fn;
            ((u32x4*)&fr)[0] = ldg_b128(wr + kk);
            ((u32x4*)&fr)[1] = ldg_b128(wr + kk + 16);
            ((u32x4*)&fz)[0] = ldg_b128(wz + kk);
            ((u32x4*)&fz)[1] = ldg_b128(wz + kk + 16);
            ((u32x4*)&fn)[0] = ldg_b128(wn + kk);
            ((u32x4*)&fn)[1] = ldg_b128(wn + kk + 16);

            cr = __builtin_amdgcn_wmma_f32_16x16x32_bf16(
                    false, a, false, fr, (short)0, cr, false, false);
            cz = __builtin_amdgcn_wmma_f32_16x16x32_bf16(
                    false, a, false, fz, (short)0, cz, false, false);
            if (kk < kH) {   // h-side of n gate (r multiplies only this part)
                cnh = __builtin_amdgcn_wmma_f32_16x16x32_bf16(
                        false, a, false, fn, (short)0, cnh, false, false);
            } else {         // x-side of n gate
                cnx = __builtin_amdgcn_wmma_f32_16x16x32_bf16(
                        false, a, false, fn, (short)0, cnx, false, false);
            }
        }

        // --- PyTorch GRU gate math, fully register-local per wave -----------
        #pragma unroll
        for (int e = 0; e < 8; ++e) {
            float rg = sigmoidf_(cr[e] + br);
            float zg = sigmoidf_(cz[e] + bz);
            float ng = tanhf((cnx[e] + bnx) + rg * (cnh[e] + bnh));
            h[e] = (1.0f - zg) * ng + zg * h[e];
        }
        __syncthreads();   // A-buffer reads done before next step's scatter
    }

    // --- head: p = sigmoid(h_last0 . w_lin + b_lin); out = [p, 1-p] ---------
    #pragma unroll
    for (int e = 0; e < 8; ++e) {
        int m = e + kh * 8;
        hfin[m * kH + col] = h[e];
    }
    __syncthreads();
    if (tid < kBTILE) {
        float acc = b_lin[0];
        for (int j = 0; j < kH; ++j) acc += hfin[tid * kH + j] * w_lin[j];
        float p = sigmoidf_(acc);
        int gr = rowbase + tid;
        out[2 * gr]     = p;
        out[2 * gr + 1] = 1.0f - p;
    }
}

// ---------------------------------------------------------------------------
// Entry point
// ---------------------------------------------------------------------------
extern "C" void kernel_launch(void* const* d_in, const int* in_sizes, int n_in,
                              void* d_out, int out_size, void* d_ws, size_t ws_size,
                              hipStream_t stream) {
    (void)in_sizes; (void)n_in; (void)out_size; (void)ws_size;
    const float* x     = (const float*)d_in[0];
    const float* w_ih0 = (const float*)d_in[1];
    const float* w_hh0 = (const float*)d_in[2];
    const float* b_ih0 = (const float*)d_in[3];
    const float* b_hh0 = (const float*)d_in[4];
    // d_in[5..8] = layer-1 params: dead w.r.t. the reference output (it only
    // uses h_last0), so layer 1 is skipped entirely.
    const float* w_lin = (const float*)d_in[9];
    const float* b_lin = (const float*)d_in[10];

    __bf16* wc = (__bf16*)d_ws;   // 768*320*2 = 480 KB of workspace

    gru_prep_weights<<<(kG3 * kK + 255) / 256, 256, 0, stream>>>(w_ih0, w_hh0, wc);
    gru0_kernel<<<kB / kBTILE, kTHREADS, 0, stream>>>(
        x, wc, b_ih0, b_hh0, w_lin, b_lin, (float*)d_out);
}
